// RGATLayer_51642686767647
// MI455X (gfx1250) — compile-verified, matrix operand order checked by reading, not run on hardware
//
#include <hip/hip_runtime.h>
#include <hip/hip_bf16.h>

#define N_NODES   100000
#define N_EDGES   1600000
#define IN_FEAT   128
#define OUT_FEAT  128
#define NUM_RELS  8
#define NUM_HEADS 8
#define HEAD_DIM  16
#define NUM_BASES 4
#define NEG_SLOPE 0.2f
#define BN_EPS    1e-5f

typedef __attribute__((ext_vector_type(16))) __bf16 v16bf;
typedef __attribute__((ext_vector_type(8)))  float  v8f;
typedef __attribute__((ext_vector_type(4)))  float  v4f;

// round-to-nearest-even f32 -> bf16 (as raw u16)
__device__ __forceinline__ unsigned short f2bf(float x) {
    union { float f; unsigned u; } v; v.f = x;
    unsigned u = v.u;
    return (unsigned short)((u + 0x7FFFu + ((u >> 16) & 1u)) >> 16);
}

// w_rel[r][k][o] through the reference's double view (basis decomposition)
__device__ __forceinline__ float wrel_val(const float* __restrict__ weight,
                                          const float* __restrict__ w_comp,
                                          int r, int k, int o) {
    int flat = r * 16384 + k * 128 + o;        // [8,128,128] flat index
    int i  = flat >> 10;                       // [128,8,128] view
    int rr = (flat >> 7) & 7;
    int oo = flat & 127;
    float s = 0.f;
#pragma unroll
    for (int b = 0; b < NUM_BASES; ++b)
        s += w_comp[rr * NUM_BASES + b] * weight[i * 512 + b * 128 + oo];
    return s;
}

// ---------------------------------------------------------------------------
// K1: build B fragments, bf16, layout [r][ct=0..8][kk][lane][e].
//   ct 0..7 : w_rel column tiles (o = ct*16 + (lane&15))
//   ct == 8 : fused attention tile: col c<8 holds
//             wl[r][k][c] = sum_hd w_rel[r][k][c*16+hd]*attn_l[c*16+hd]; c>=8 -> 0
//   element e = B[k][col], k = kk*32 + 8*(lane>>4) + (e<8?e:e+8)
// ---------------------------------------------------------------------------
__global__ void k_wrel(const float* __restrict__ weight,   // [4,128,128] flat
                       const float* __restrict__ w_comp,   // [8,4]
                       const float* __restrict__ attn_l,   // [8,16] flat
                       unsigned short* __restrict__ bbuf) {
    int j    = blockIdx.x * 256 + threadIdx.x;            // 147456 total
    int e    = j & 15;
    int lane = (j >> 4) & 31;
    int kk   = (j >> 9) & 3;
    int rc   = j >> 11;
    int ct   = rc % 9;
    int r    = rc / 9;
    int k = kk * 32 + ((lane >> 4) << 3) + (e < 8 ? e : e + 8);
    int c = lane & 15;
    float s = 0.f;
    if (ct < 8) {
        s = wrel_val(weight, w_comp, r, k, ct * 16 + c);
    } else if (c < NUM_HEADS) {
#pragma unroll
        for (int hd = 0; hd < HEAD_DIM; ++hd)
            s += wrel_val(weight, w_comp, r, k, c * 16 + hd) * attn_l[c * 16 + hd];
    }
    bbuf[j] = f2bf(s);
}

// ---------------------------------------------------------------------------
// K2a: h -> bf16 pre-swizzled into WMMA A-fragment order
// ---------------------------------------------------------------------------
__global__ void k_swizzle_h(const float* __restrict__ h,
                            unsigned short* __restrict__ abuf) {
    int j    = blockIdx.x * 256 + threadIdx.x;   // 12.8M total
    int e    = j & 15;
    int lane = (j >> 4) & 31;
    int kk   = (j >> 9) & 3;
    int tile = j >> 11;
    int node = tile * 16 + (lane & 15);
    int k    = kk * 32 + ((lane >> 4) << 3) + (e < 8 ? e : e + 8);
    abuf[j] = f2bf(h[(size_t)node * IN_FEAT + k]);
}

// ---------------------------------------------------------------------------
// K2b: per-node right attention scores r_score[n][head]
// ---------------------------------------------------------------------------
__global__ void k_rscore(const float* __restrict__ h,
                         const float* __restrict__ attn_r,   // [8,16] flat
                         float* __restrict__ rsc) {
    __shared__ float sm[256];
    int t    = threadIdx.x;
    int node = blockIdx.x * 2 + (t >> 7);
    int c    = t & 127;
    sm[t] = h[(size_t)node * IN_FEAT + c] * attn_r[c];
    __syncthreads();
    if ((c & 15) == 0) {
        float s = 0.f;
#pragma unroll
        for (int q = 0; q < 16; ++q) s += sm[(t & ~15) + q];
        rsc[node * NUM_HEADS + (c >> 4)] = s;
    }
}

// ---------------------------------------------------------------------------
// K3: WMMA GEMM. One block = 80 nodes (5 sub-tiles of 16) x 8 rels.
// A fragments for all 5 sub-tiles stay in registers; B fragments loaded once
// per (r,ct) and reused across sub-tiles. ct 0..7 -> xh (non-temporal stores);
// ct == 8 -> l_score directly from the D layout (no cross-lane ops at all).
// 180 v_wmma_f32_16x16x32_bf16 per wave.
// ---------------------------------------------------------------------------
__global__ void __launch_bounds__(256)
k_gemm(const unsigned short* __restrict__ abuf,
       const unsigned short* __restrict__ bbuf,
       float* __restrict__ xh,                         // [N,8,128] f32
       float* __restrict__ lsc) {                      // [N,8,8]  f32
    int tile0 = blockIdx.x * 5;                        // 1250 blocks x 5 tiles
    int r     = threadIdx.x >> 5;                      // relation = wave id
    int lane  = threadIdx.x & 31;
    int m_hi  = (lane >> 4) << 3;
    int n     = lane & 15;

    v16bf a[5][4];
#pragma unroll
    for (int g = 0; g < 5; ++g) {
        const v16bf* ap = (const v16bf*)(abuf + (size_t)(tile0 + g) * 4 * 32 * 16);
#pragma unroll
        for (int kk = 0; kk < 4; ++kk) a[g][kk] = ap[kk * 32 + lane];
    }

#pragma unroll
    for (int ct = 0; ct < 9; ++ct) {
        const v16bf* bp = (const v16bf*)(bbuf + (size_t)((r * 9 + ct) * 4) * 32 * 16);
        v16bf b0 = bp[0 * 32 + lane];
        v16bf b1 = bp[1 * 32 + lane];
        v16bf b2 = bp[2 * 32 + lane];
        v16bf b3 = bp[3 * 32 + lane];

#pragma unroll
        for (int g = 0; g < 5; ++g) {
            v8f acc = {};
            acc = __builtin_amdgcn_wmma_f32_16x16x32_bf16(false, a[g][0], false, b0, (short)0, acc, false, false);
            acc = __builtin_amdgcn_wmma_f32_16x16x32_bf16(false, a[g][1], false, b1, (short)0, acc, false, false);
            acc = __builtin_amdgcn_wmma_f32_16x16x32_bf16(false, a[g][2], false, b2, (short)0, acc, false, false);
            acc = __builtin_amdgcn_wmma_f32_16x16x32_bf16(false, a[g][3], false, b3, (short)0, acc, false, false);

            int node0 = (tile0 + g) * 16 + m_hi;
            if (ct < 8) {
#pragma unroll
                for (int i = 0; i < 8; ++i)
                    __builtin_nontemporal_store(
                        acc[i], xh + (size_t)(node0 + i) * 1024 + r * 128 + ct * 16 + n);
            } else if (n < NUM_HEADS) {
#pragma unroll
                for (int i = 0; i < 8; ++i)
                    lsc[(size_t)(node0 + i) * 64 + r * 8 + n] = acc[i];
            }
        }
    }
}

// ---------------------------------------------------------------------------
// K4: edge kernel. One wave per edge; lanes 0..7 build head-softmax, all
// lanes apply per-head weights to 4 features each and atomically accumulate.
// xh gather is non-temporal (streamed once); agg stays L2-resident.
// ---------------------------------------------------------------------------
__global__ void k_edge(const int* __restrict__ src, const int* __restrict__ dst,
                       const int* __restrict__ rel,
                       const float* __restrict__ lsc, const float* __restrict__ rsc,
                       const float* __restrict__ xh, float* __restrict__ agg) {
    int e    = blockIdx.x * 8 + (threadIdx.x >> 5);    // 1.6M edges exactly
    int lane = threadIdx.x & 31;
    int s = src[e], d = dst[e], r = rel[e];

    float x = 0.f;
    if (lane < 8) {
        float v = lsc[(size_t)s * 64 + r * 8 + lane] + rsc[d * 8 + lane];
        x = v > 0.f ? v : NEG_SLOPE * v;               // leaky_relu
    }
    float mx = x;
    mx = fmaxf(mx, __shfl_xor(mx, 1, 8));
    mx = fmaxf(mx, __shfl_xor(mx, 2, 8));
    mx = fmaxf(mx, __shfl_xor(mx, 4, 8));
    float ex  = __expf(x - mx);
    float sum = ex;
    sum += __shfl_xor(sum, 1, 8);
    sum += __shfl_xor(sum, 2, 8);
    sum += __shfl_xor(sum, 4, 8);
    float p  = ex / sum;                               // valid in lanes 0..7
    float ph = __shfl(p, lane >> 2, 32);               // head of my 4 features

    v4f v = __builtin_nontemporal_load(
        (const v4f*)(xh + (size_t)s * 1024 + (size_t)r * 128 + lane * 4));
    float* ag = agg + (size_t)d * 128 + lane * 4;
    unsafeAtomicAdd(ag + 0, v.x * ph);
    unsafeAtomicAdd(ag + 1, v.y * ph);
    unsafeAtomicAdd(ag + 2, v.z * ph);
    unsafeAtomicAdd(ag + 3, v.w * ph);
}

// ---------------------------------------------------------------------------
// K5: per-channel sum / sumsq over nodes (batch-norm stats)
// ---------------------------------------------------------------------------
__global__ void k_bnstat(const float* __restrict__ agg, float* __restrict__ stats) {
    __shared__ float s1[256], s2[256];
    int t = threadIdx.x;
    int c = t & 127, half = t >> 7;
    float sum = 0.f, sq = 0.f;
    for (int row = blockIdx.x * 2 + half; row < N_NODES; row += gridDim.x * 2) {
        float v = agg[(size_t)row * 128 + c];
        sum += v; sq += v * v;
    }
    s1[t] = sum; s2[t] = sq;
    __syncthreads();
    if (t < 128) {
        unsafeAtomicAdd(&stats[t],       s1[t] + s1[t + 128]);
        unsafeAtomicAdd(&stats[128 + t], s2[t] + s2[t + 128]);
    }
}

// ---------------------------------------------------------------------------
// K6: batch-norm apply (biased variance = E[x^2] - mean^2)
// ---------------------------------------------------------------------------
__global__ void k_bnapply(const float* __restrict__ agg, const float* __restrict__ stats,
                          const float* __restrict__ gamma, const float* __restrict__ beta,
                          float* __restrict__ out) {
    size_t j = (size_t)blockIdx.x * 256 + threadIdx.x;
    int c = (int)(j & 127);
    const float inv_n = 1.0f / (float)N_NODES;
    float mean = stats[c] * inv_n;
    float var  = stats[128 + c] * inv_n - mean * mean;
    out[j] = (agg[j] - mean) * rsqrtf(var + BN_EPS) * gamma[c] + beta[c];
}

// ---------------------------------------------------------------------------
// launcher
// ---------------------------------------------------------------------------
extern "C" void kernel_launch(void* const* d_in, const int* in_sizes, int n_in,
                              void* d_out, int out_size, void* d_ws, size_t ws_size,
                              hipStream_t stream) {
    const float* h      = (const float*)d_in[0];
    const float* weight = (const float*)d_in[1];
    const float* w_comp = (const float*)d_in[2];
    const float* attn_l = (const float*)d_in[3];
    const float* attn_r = (const float*)d_in[4];
    const float* gamma  = (const float*)d_in[5];
    const float* beta   = (const float*)d_in[6];
    const int*   src    = (const int*)d_in[7];
    const int*   dst    = (const int*)d_in[8];
    const int*   rel    = (const int*)d_in[9];
    float* out = (float*)d_out;

    char* ws = (char*)d_ws;
    const size_t OFF_BBUF  = 0;                               //    294,912 B (bf16, 9 ct-tiles)
    const size_t OFF_ABUF  = 294912;                          // 25,600,000 B
    const size_t OFF_LSC   = 25894912;                        // 25,600,000 B
    const size_t OFF_RSC   = 51494912;                        //  3,200,000 B
    const size_t OFF_AGG   = 54694912;                        // 51,200,000 B
    const size_t OFF_STAT  = 105894912;                       //      1,024 B
    const size_t OFF_XH    = 105895936;                       // 409,600,000 B

    unsigned short* bbuf = (unsigned short*)(ws + OFF_BBUF);
    unsigned short* abuf = (unsigned short*)(ws + OFF_ABUF);
    float* lsc   = (float*)(ws + OFF_LSC);
    float* rsc   = (float*)(ws + OFF_RSC);
    float* agg   = (float*)(ws + OFF_AGG);
    float* stats = (float*)(ws + OFF_STAT);
    float* xh    = (float*)(ws + OFF_XH);

    hipMemsetAsync(agg, 0, 51200000 + 1024, stream);          // agg+stats contiguous

    k_wrel     <<<576,    256, 0, stream>>>(weight, w_comp, attn_l, bbuf);
    k_swizzle_h<<<50000,  256, 0, stream>>>(h, abuf);
    k_rscore   <<<50000,  256, 0, stream>>>(h, attn_r, rsc);
    k_gemm     <<<1250,   256, 0, stream>>>(abuf, bbuf, xh, lsc);
    k_edge     <<<200000, 256, 0, stream>>>(src, dst, rel, lsc, rsc, xh, agg);
    k_bnstat   <<<512,    256, 0, stream>>>(agg, stats);
    k_bnapply  <<<50000,  256, 0, stream>>>(agg, stats, gamma, beta, out);
}
